// GATNet_38508676776213
// MI455X (gfx1250) — matrix operand order repaired and checked by reading, not compiled, not run on hardware
//
#include <hip/hip_runtime.h>
#include <hip/hip_bf16.h>
#include <math.h>

typedef __attribute__((ext_vector_type(16))) _Float16 v16h;
typedef __attribute__((ext_vector_type(8)))  _Float16 v8h;
typedef __attribute__((ext_vector_type(8)))  float    v8f;

#define NNODES 100000
#define NEDGES 1600000
#define NGRAPH 64
#define SLOPE  0.2f
#define MTILES ((NNODES + 15) / 16)   // 6250

// ---- monotone float<->uint mapping so atomicMax(uint) == float max ----
__device__ __forceinline__ unsigned fmap(float f) {
  unsigned b = __float_as_uint(f);
  return (b & 0x80000000u) ? ~b : (b | 0x80000000u);
}
__device__ __forceinline__ float funmap(unsigned u) {
  unsigned b = (u & 0x80000000u) ? (u & 0x7FFFFFFFu) : ~u;
  return __uint_as_float(b);
}

// ============================================================================
// WMMA GEMM: D[N, NC] = A[N, K] * B[K, NC], f16 inputs (converted on load),
// f32 accumulate.
//  - 256-thread block = 8 waves; each wave owns one 16x16 output tile in a
//    128-row strip (blockIdx.x), column block n0 (blockIdx.y).
//  - W tile staged TRANSPOSED in LDS (WT[col][k], rows padded by 8 halves =>
//    16B-aligned, bank-spread): each lane's B fragment is 16 contiguous
//    halves -> two ds_load_b128 instead of 16 ds_load_u16.
//  - K templated so the K-loop fully unrolls into back-to-back v_wmma ops.
// VGPR layouts: 16-bit A/B use contiguous k order (k = 16h + s); f32 C/D 16x16 per CDNA5 ISA 7.12.2.
// ============================================================================
template <int K, int NC>
__global__ __launch_bounds__(256) void gat_gemm_f16(
    const float* __restrict__ A, const float* __restrict__ B,
    float* __restrict__ D) {
  constexpr int KP = K + 8;                   // row stride (halves), 16B aligned
  __shared__ __align__(32) _Float16 WT[16][KP];

  const int n0 = blockIdx.y << 4;
  // cooperative stage of the W column tile (f32 -> f16, transposed,
  // zero-pad cols >= NC); global reads coalesced over columns.
  for (int idx = threadIdx.x; idx < K * 16; idx += 256) {
    int k = idx >> 4, c = idx & 15;
    int col = n0 + c;
    WT[c][k] = (col < NC) ? (_Float16)B[(size_t)k * NC + col] : (_Float16)0.0f;
  }
  __syncthreads();

  const int wave = threadIdx.x >> 5;
  const int lane = threadIdx.x & 31;
  const int half = lane >> 4;       // 0: lanes 0-15, 1: lanes 16-31
  const int l    = lane & 15;
  const int tile = blockIdx.x * 8 + wave;
  if (tile >= MTILES) return;       // tail strip: some waves idle (after barrier)
  const int m0 = tile << 4;

  v8f acc = {};
  const float*    arow = A + (size_t)(m0 + l) * K + half * 16;
  const _Float16* bcol = &WT[l][half * 16];   // 16B aligned: l*KP*2 + half*32

#pragma unroll
  for (int k0 = 0; k0 < K; k0 += 32) {
    // A 16x32 f16 fragment: lane<16 holds M=l, K = k0+{0..15};
    // lanes 16-31 hold k0+{16..31} (matches B's contiguous order).
    v16h a;
#pragma unroll
    for (int j = 0; j < 8; ++j) {
      a[j]     = (_Float16)arow[k0 + j];
      a[8 + j] = (_Float16)arow[k0 + 8 + j];
    }
    // B 32x16 f16 fragment: lanes 0-15 hold K=k0+{0..15} of column N=l;
    // lanes 16-31 hold k0+{16..31}. Contiguous in WT -> 2x ds_load_b128.
    v8h b0 = *(const v8h*)(bcol + k0);
    v8h b1 = *(const v8h*)(bcol + k0 + 8);
    v16h b;
#pragma unroll
    for (int j = 0; j < 8; ++j) { b[j] = b0[j]; b[8 + j] = b1[j]; }

    acc = __builtin_amdgcn_wmma_f32_16x16x32_f16(
        /*neg_a=*/false, a, /*neg_b=*/false, b,
        /*c_mod=*/(short)0, acc, /*reuse_a=*/false, /*reuse_b=*/false);
  }

  // D 16x16 f32: VGPR r -> row = m0 + r + half*8, col = n0 + l
  const int col = n0 + l;
  if (col < NC) {
#pragma unroll
    for (int r = 0; r < 8; ++r)
      D[(size_t)(m0 + r + half * 8) * NC + col] = acc[r];
  }
}

// ---- per-(node, head) attention logits: s = sum_c h[n,h,c] * a[h,c] ----
__global__ void gat_node_logits(const float* __restrict__ h,
                                const float* __restrict__ a_src,
                                const float* __restrict__ a_dst,
                                float* __restrict__ ssrc, float* __restrict__ sdst,
                                int H, int C) {
  int t = blockIdx.x * blockDim.x + threadIdx.x;
  if (t >= NNODES * H) return;
  int n = t / H, hh = t - n * H;
  const float* hp = h + (size_t)n * H * C + (size_t)hh * C;
  const float* as = a_src + hh * C;
  const float* ad = a_dst + hh * C;
  float s0 = 0.f, s1 = 0.f;
  for (int c = 0; c < C; ++c) { float v = hp[c]; s0 += v * as[c]; s1 += v * ad[c]; }
  ssrc[t] = s0;
  sdst[t] = s1;
}

// ---- segment max over incoming edges (incl. self loops) ----
__global__ void gat_edge_amax(const int* __restrict__ ei,
                              const float* __restrict__ ssrc,
                              const float* __restrict__ sdst,
                              unsigned* __restrict__ amax, int H) {
  long long t = (long long)blockIdx.x * blockDim.x + threadIdx.x;
  const long long ET = (long long)(NEDGES + NNODES) * H;
  if (t >= ET) return;
  int e = (int)(t / H), hh = (int)(t - (long long)e * H);
  int s = (e < NEDGES) ? ei[e] : (e - NEDGES);
  int d = (e < NEDGES) ? ei[NEDGES + e] : (e - NEDGES);
  float v = ssrc[s * H + hh] + sdst[d * H + hh];
  v = (v > 0.f) ? v : SLOPE * v;
  atomicMax(&amax[d * H + hh], fmap(v));
}

// ---- segment sum of exp(logit - max) ----
__global__ void gat_edge_denom(const int* __restrict__ ei,
                               const float* __restrict__ ssrc,
                               const float* __restrict__ sdst,
                               const unsigned* __restrict__ amax,
                               float* __restrict__ denom, int H) {
  long long t = (long long)blockIdx.x * blockDim.x + threadIdx.x;
  const long long ET = (long long)(NEDGES + NNODES) * H;
  if (t >= ET) return;
  int e = (int)(t / H), hh = (int)(t - (long long)e * H);
  int s = (e < NEDGES) ? ei[e] : (e - NEDGES);
  int d = (e < NEDGES) ? ei[NEDGES + e] : (e - NEDGES);
  float v = ssrc[s * H + hh] + sdst[d * H + hh];
  v = (v > 0.f) ? v : SLOPE * v;
  atomicAdd(&denom[d * H + hh], __expf(v - funmap(amax[d * H + hh])));
}

// ---- weighted scatter-add aggregation: out[dst] += alpha * h[src] ----
// One thread per (edge, 4 channels).
__global__ void gat_edge_aggr(const int* __restrict__ ei,
                              const float* __restrict__ ssrc,
                              const float* __restrict__ sdst,
                              const unsigned* __restrict__ amax,
                              const float* __restrict__ denom,
                              const float* __restrict__ h,
                              float* __restrict__ out, int H, int C) {
  const int HC4 = (H * C) >> 2;
  long long t = (long long)blockIdx.x * blockDim.x + threadIdx.x;
  const long long TT = (long long)(NEDGES + NNODES) * HC4;
  if (t >= TT) return;
  int e = (int)(t / HC4), q = (int)(t - (long long)e * HC4);
  int s = (e < NEDGES) ? ei[e] : (e - NEDGES);
  int d = (e < NEDGES) ? ei[NEDGES + e] : (e - NEDGES);
  int hh = (q << 2) / C;
  float v = ssrc[s * H + hh] + sdst[d * H + hh];
  v = (v > 0.f) ? v : SLOPE * v;
  float w = __expf(v - funmap(amax[d * H + hh])) / denom[d * H + hh];
  const float4 hv = *(const float4*)(h + (size_t)s * H * C + (q << 2));
  float* op = out + (size_t)d * H * C + (q << 2);
  atomicAdd(op + 0, w * hv.x);
  atomicAdd(op + 1, w * hv.y);
  atomicAdd(op + 2, w * hv.z);
  atomicAdd(op + 3, w * hv.w);
}

// ---- bias (+ optional ReLU), in place ----
__global__ void gat_bias_act(float* __restrict__ x, const float* __restrict__ bias,
                             int HC, int do_relu) {
  long long t = (long long)blockIdx.x * blockDim.x + threadIdx.x;
  if (t >= (long long)NNODES * HC) return;
  int j = (int)(t % HC);
  float v = x[t] + bias[j];
  if (do_relu) v = (v > 0.f) ? v : 0.f;
  x[t] = v;
}

// ---- global mean pool accumulation ----
__global__ void gat_pool(const float* __restrict__ h3, const int* __restrict__ batch,
                         float* __restrict__ pooled, float* __restrict__ cnt) {
  int t = blockIdx.x * blockDim.x + threadIdx.x;
  if (t >= NNODES * 8) return;
  int n = t >> 3, j = t & 7;
  int g = batch[n];
  atomicAdd(&pooled[g * 8 + j], h3[t]);
  if (j == 0) atomicAdd(&cnt[g], 1.0f);
}

__global__ void gat_final(const float* __restrict__ pooled,
                          const float* __restrict__ cnt,
                          float* __restrict__ out) {
  int t = blockIdx.x * blockDim.x + threadIdx.x;
  if (t >= NGRAPH * 8) return;
  float c = cnt[t >> 3];
  float v = pooled[t] / fmaxf(c, 1.0f);
  out[t] = 1.0f / (1.0f + __expf(-v));
}

// ============================================================================
struct LayerBufs {
  const int* ei;
  float *ssrc, *sdst, *denom;
  unsigned* amax;
  hipStream_t stream;
};

template <int K, int H, int C>
static void run_layer(const LayerBufs& lb, const float* xin, const float* W,
                      const float* as, const float* ad, const float* bias,
                      float* hbuf, float* aggbuf, int do_relu) {
  constexpr int HC = H * C;
  // dense transform (WMMA, LDS-staged transposed weights)
  dim3 grid((MTILES + 7) / 8, (HC + 15) / 16);
  gat_gemm_f16<K, HC><<<grid, dim3(256), 0, lb.stream>>>(xin, W, hbuf);
  // attention logits
  int nt = NNODES * H;
  gat_node_logits<<<(nt + 255) / 256, 256, 0, lb.stream>>>(hbuf, as, ad,
                                                           lb.ssrc, lb.sdst, H, C);
  // init segment buffers / aggregation target
  hipMemsetAsync(lb.amax, 0, (size_t)NNODES * H * sizeof(unsigned), lb.stream);
  hipMemsetAsync(lb.denom, 0, (size_t)NNODES * H * sizeof(float), lb.stream);
  hipMemsetAsync(aggbuf, 0, (size_t)NNODES * HC * sizeof(float), lb.stream);
  long long et = (long long)(NEDGES + NNODES) * H;
  gat_edge_amax<<<(unsigned)((et + 255) / 256), 256, 0, lb.stream>>>(
      lb.ei, lb.ssrc, lb.sdst, lb.amax, H);
  gat_edge_denom<<<(unsigned)((et + 255) / 256), 256, 0, lb.stream>>>(
      lb.ei, lb.ssrc, lb.sdst, lb.amax, lb.denom, H);
  long long ta = (long long)(NEDGES + NNODES) * (HC / 4);
  gat_edge_aggr<<<(unsigned)((ta + 255) / 256), 256, 0, lb.stream>>>(
      lb.ei, lb.ssrc, lb.sdst, lb.amax, lb.denom, hbuf, aggbuf, H, C);
  long long tb = (long long)NNODES * HC;
  gat_bias_act<<<(unsigned)((tb + 255) / 256), 256, 0, lb.stream>>>(aggbuf, bias,
                                                                    HC, do_relu);
}

extern "C" void kernel_launch(void* const* d_in, const int* in_sizes, int n_in,
                              void* d_out, int out_size, void* d_ws, size_t ws_size,
                              hipStream_t stream) {
  const float* x     = (const float*)d_in[0];
  const int*   ei    = (const int*)d_in[1];
  const int*   batch = (const int*)d_in[2];
  const float* W1  = (const float*)d_in[3];
  const float* as1 = (const float*)d_in[4];
  const float* ad1 = (const float*)d_in[5];
  const float* b1  = (const float*)d_in[6];
  const float* W2  = (const float*)d_in[7];
  const float* as2 = (const float*)d_in[8];
  const float* ad2 = (const float*)d_in[9];
  const float* b2  = (const float*)d_in[10];
  const float* W3  = (const float*)d_in[11];
  const float* as3 = (const float*)d_in[12];
  const float* ad3 = (const float*)d_in[13];
  const float* b3  = (const float*)d_in[14];
  float* out = (float*)d_out;

  // workspace layout (floats)
  float* B0 = (float*)d_ws;
  float* B1 = B0 + (size_t)NNODES * 128;
  float* B2 = B1 + (size_t)NNODES * 128;
  float* ssrc = B2 + (size_t)NNODES * 128;
  float* sdst = ssrc + (size_t)NNODES * 4;
  unsigned* amax = (unsigned*)(sdst + (size_t)NNODES * 4);
  float* denom = (float*)amax + (size_t)NNODES * 4;
  float* pooled = denom + (size_t)NNODES * 4;
  float* cnt = pooled + NGRAPH * 8;

  LayerBufs lb{ei, ssrc, sdst, denom, amax, stream};

  // layer 1: [N,128] -> 4 heads x 16 -> [N,64], relu
  run_layer<128, 4, 16>(lb, x, W1, as1, ad1, b1, B0, B1, 1);
  // layer 2: [N,64] -> 4 heads x 32 -> [N,128], relu
  run_layer<64, 4, 32>(lb, B1, W2, as2, ad2, b2, B0, B2, 1);
  // layer 3: [N,128] -> 1 head x 8 -> [N,8], no relu
  run_layer<128, 1, 8>(lb, B2, W3, as3, ad3, b3, B1, B0, 0);

  // global mean pool per graph + sigmoid
  hipMemsetAsync(pooled, 0, (NGRAPH * 8 + NGRAPH) * sizeof(float), stream);
  gat_pool<<<(NNODES * 8 + 255) / 256, 256, 0, stream>>>(B0, batch, pooled, cnt);
  gat_final<<<1, 512, 0, stream>>>(pooled, cnt, out);
}
